// NodeAttention_40613210751278
// MI455X (gfx1250) — matrix operand order repaired and checked by reading, not compiled, not run on hardware
//
#include <hip/hip_runtime.h>
#include <hip/hip_bf16.h>
#include <math.h>

typedef __bf16 bf16;
typedef __attribute__((ext_vector_type(16))) __bf16 v16bf;
typedef __attribute__((ext_vector_type(4)))  __bf16 v4bf;
typedef __attribute__((ext_vector_type(8)))  float  v8f;

// ---------------------------------------------------------------------------
// fp32 -> bf16 conversion (vectorized by 4)
// ---------------------------------------------------------------------------
__global__ __launch_bounds__(256) void cvt_f32_bf16_kernel(
    const float4* __restrict__ in, v4bf* __restrict__ out, int n4)
{
    int i = blockIdx.x * blockDim.x + threadIdx.x;
    if (i < n4) {
        float4 f = in[i];
        v4bf b;
        b[0] = (bf16)f.x; b[1] = (bf16)f.y; b[2] = (bf16)f.z; b[3] = (bf16)f.w;
        out[i] = b;
    }
}

// ---------------------------------------------------------------------------
// Tiled WMMA GEMM (NT only):  C[Mrows x Ncols] = A[Mrows x K] * B[Ncols x K]^T
//   A32 = true : A is fp32, converted to bf16 during (VGPR) staging;
//                otherwise A is bf16 and staged with async global->LDS copies.
//   B is always bf16, staged with GLOBAL_LOAD_ASYNC_TO_LDS_B128.
//   OUT = 0: fp32*alpha, 1: bf16, 2: elu(fp32), 3: bf16 + bf16-transposed
// Block: 256 threads (8 waves). Tile BM=BN=128, BK=32, double-buffered LDS.
// Wave (wr,wc) in 2x4 grid computes a 64x32 sub-tile = 4x2 WMMA accumulators.
// ---------------------------------------------------------------------------
constexpr int BM = 128, BN = 128, BK = 32, LDT = BK + 8; // LDS row stride (elems)

union FragU { v16bf v; unsigned u[8]; };

template <bool A32, int OUT>
__global__ __launch_bounds__(256) void gemm_wmma_kernel(
    const void* __restrict__ Aptr, const bf16* __restrict__ Bptr,
    float* __restrict__ Cf, bf16* __restrict__ Cb, bf16* __restrict__ CbT,
    int Mrows, int Ncols, int K, float alpha)
{
    __shared__ alignas(16) bf16 As[2][BM * LDT];
    __shared__ alignas(16) bf16 Bs[2][BN * LDT];

    const int tid  = threadIdx.x;
    const int lane = tid & 31;
    const int wave = tid >> 5;
    const int half = lane >> 4;   // 0 or 1
    const int ln   = lane & 15;
    const int wr   = wave >> 2;   // 0..1
    const int wc   = wave & 3;    // 0..3

    const int nTilesN = Ncols / BN;
    const int bm0 = (blockIdx.x / nTilesN) * BM;
    const int bn0 = (blockIdx.x % nTilesN) * BN;

    v8f acc[4][2];
#pragma unroll
    for (int i = 0; i < 4; ++i)
#pragma unroll
        for (int j = 0; j < 2; ++j) acc[i][j] = 0.0f;

    // ---- tile staging: B (and A when bf16) via async global->LDS copies ----
    auto stage = [&](int buf, int k0) {
#pragma unroll
        for (int q = 0; q < 2; ++q) {
            int chunk = q * 256 + tid;        // 0..511, 8 bf16 (16B) each
            int r  = chunk >> 2;
            int kc = (chunk & 3) * 8;
            unsigned ldsB = (unsigned)(size_t)(const void*)&Bs[buf][r * LDT + kc];
            const bf16* gB = Bptr + (size_t)(bn0 + r) * K + k0 + kc;
            asm volatile("global_load_async_to_lds_b128 %0, %1, off"
                         :: "v"(ldsB), "v"(gB) : "memory");
        }
        if constexpr (!A32) {
            const bf16* A = (const bf16*)Aptr;
#pragma unroll
            for (int q = 0; q < 2; ++q) {
                int chunk = q * 256 + tid;
                int r  = chunk >> 2;
                int kc = (chunk & 3) * 8;
                unsigned ldsA = (unsigned)(size_t)(const void*)&As[buf][r * LDT + kc];
                const bf16* gA = A + (size_t)(bm0 + r) * K + k0 + kc;
                asm volatile("global_load_async_to_lds_b128 %0, %1, off"
                             :: "v"(ldsA), "v"(gA) : "memory");
            }
        } else {
            const float* A = (const float*)Aptr;
#pragma unroll
            for (int q = 0; q < 2; ++q) {
                int chunk = q * 256 + tid;
                int r  = chunk >> 2;
                int kc = (chunk & 3) * 8;
                const float* g = A + (size_t)(bm0 + r) * K + k0 + kc;
                bf16* s = &As[buf][r * LDT + kc];
#pragma unroll
                for (int j = 0; j < 8; ++j) s[j] = (bf16)g[j];
            }
        }
    };

    stage(0, 0);
    asm volatile("s_wait_asynccnt 0x0" ::: "memory");
    __syncthreads();

    int buf = 0;
    for (int k0 = 0; k0 < K; k0 += BK) {
        if (k0 + BK < K) stage(buf ^ 1, k0 + BK);   // overlap with compute

        // ---- per-lane fragment gathers (ISA 7.12.2 16-bit layouts) ----
        FragU af[4], bfr[2];
#pragma unroll
        for (int i = 0; i < 4; ++i) {
            const unsigned short* base =
                (const unsigned short*)&As[buf][(wr * 64 + i * 16 + ln) * LDT];
#pragma unroll
            for (int p = 0; p < 8; ++p) {
                int k = (p < 4 ? 2 * p : 2 * p + 8) + 8 * half;
                af[i].u[p] = *(const unsigned*)(base + k);
            }
        }
#pragma unroll
        for (int j = 0; j < 2; ++j) {
            const unsigned short* base =
                (const unsigned short*)&Bs[buf][(wc * 32 + j * 16 + ln) * LDT];
#pragma unroll
            for (int p = 0; p < 8; ++p) {
                int k = (p < 4 ? 2 * p : 2 * p + 8) + 8 * half;
                bfr[j].u[p] = *(const unsigned*)(base + k);
            }
        }
#pragma unroll
        for (int i = 0; i < 4; ++i)
#pragma unroll
            for (int j = 0; j < 2; ++j)
                acc[i][j] = __builtin_amdgcn_wmma_f32_16x16x32_bf16(
                    false, af[i].v, false, bfr[j].v,
                    (short)0, acc[i][j], false, false);

        asm volatile("s_wait_asynccnt 0x0" ::: "memory");
        __syncthreads();
        buf ^= 1;
    }

    // ---- epilogue ----
#pragma unroll
    for (int i = 0; i < 4; ++i)
#pragma unroll
        for (int j = 0; j < 2; ++j)
#pragma unroll
            for (int r = 0; r < 8; ++r) {
                int m = bm0 + wr * 64 + i * 16 + r + 8 * half;
                int n = bn0 + wc * 32 + j * 16 + ln;
                float v = acc[i][j][r] * alpha;
                if constexpr (OUT == 2) v = v > 0.0f ? v : expf(v) - 1.0f;
                if constexpr (OUT == 1 || OUT == 3)
                    Cb[(size_t)m * Ncols + n] = (bf16)v;
                if constexpr (OUT == 3)
                    CbT[(size_t)n * Mrows + m] = (bf16)v;
                if constexpr (OUT == 0 || OUT == 2)
                    Cf[(size_t)m * Ncols + n] = v;
            }
}

// ---------------------------------------------------------------------------
// In-place masked row softmax over M=4096 columns, single pass over S and H.
//   v = mask ? S*scale : -1e9 ; softmax ; masked lanes -> exact 0
// One 256-thread block per row; 16 contiguous elements per thread in registers.
// ---------------------------------------------------------------------------
__global__ __launch_bounds__(256) void softmax_mask_kernel(
    float* __restrict__ S, const int* __restrict__ H, float scale)
{
    __shared__ float red[256];
    const int t = threadIdx.x;
    const size_t base = (size_t)blockIdx.x * 4096 + (size_t)t * 16;

    float v[16];
    float lmax = -3.4028235e38f;
#pragma unroll
    for (int i = 0; i < 16; ++i) {
        float s = S[base + i] * scale;
        v[i] = (H[base + i] > 0) ? s : -1e9f;
        lmax = fmaxf(lmax, v[i]);
    }
    red[t] = lmax;
    __syncthreads();
    for (int s = 128; s > 0; s >>= 1) {
        if (t < s) red[t] = fmaxf(red[t], red[t + s]);
        __syncthreads();
    }
    const float rmax = red[0];
    __syncthreads();

    float lsum = 0.0f;
#pragma unroll
    for (int i = 0; i < 16; ++i) {
        float e = (v[i] > -5e8f) ? __expf(v[i] - rmax) : 0.0f;  // masked -> exact 0
        v[i] = e;
        lsum += e;
    }
    red[t] = lsum;
    __syncthreads();
    for (int s = 128; s > 0; s >>= 1) {
        if (t < s) red[t] += red[t + s];
        __syncthreads();
    }
    const float rinv = 1.0f / red[0];
#pragma unroll
    for (int i = 0; i < 16; ++i) S[base + i] = v[i] * rinv;
}

// ---------------------------------------------------------------------------
// Launch: he = e@We^T (+he^T) ; hn = x@Wn^T ; S = hn@he^T ; softmax ;
//         out = elu(S @ (he^T)^T)
// d_out = [ out (N*DIM fp32) | B (N*M fp32) ]
// ---------------------------------------------------------------------------
extern "C" void kernel_launch(void* const* d_in, const int* in_sizes, int n_in,
                              void* d_out, int out_size, void* d_ws, size_t ws_size,
                              hipStream_t stream)
{
    (void)in_sizes; (void)n_in; (void)out_size; (void)ws_size;

    constexpr int N = 16384, M = 4096, D = 1024;

    const float* x  = (const float*)d_in[0];
    const float* er = (const float*)d_in[1];
    const int*   H  = (const int*)  d_in[2];
    const float* We = (const float*)d_in[3];
    const float* Wn = (const float*)d_in[4];

    bf16* ws = (bf16*)d_ws;
    size_t o = 0;
    bf16* x_b   = ws + o; o += (size_t)N * D;
    bf16* e_b   = ws + o; o += (size_t)M * D;
    bf16* We_b  = ws + o; o += (size_t)D * D;
    bf16* Wn_b  = ws + o; o += (size_t)D * D;
    bf16* he_b  = ws + o; o += (size_t)M * D;   // [M, D]
    bf16* heT_b = ws + o; o += (size_t)D * M;   // [D, M]
    bf16* hn_b  = ws + o; o += (size_t)N * D;

    float* outp = (float*)d_out;                 // [N, D]
    float* Bmat = outp + (size_t)N * D;          // [N, M] — raw S, softmax in place

    // fp32 -> bf16 conversions
    {
        int n4;
        n4 = (N * D) / 4;
        cvt_f32_bf16_kernel<<<(n4 + 255) / 256, 256, 0, stream>>>((const float4*)x,  (v4bf*)x_b,  n4);
        n4 = (M * D) / 4;
        cvt_f32_bf16_kernel<<<(n4 + 255) / 256, 256, 0, stream>>>((const float4*)er, (v4bf*)e_b,  n4);
        n4 = (D * D) / 4;
        cvt_f32_bf16_kernel<<<(n4 + 255) / 256, 256, 0, stream>>>((const float4*)We, (v4bf*)We_b, n4);
        cvt_f32_bf16_kernel<<<(n4 + 255) / 256, 256, 0, stream>>>((const float4*)Wn, (v4bf*)Wn_b, n4);
    }

    // he[M,D] = e @ We^T   -> bf16 he_b and transposed heT_b
    gemm_wmma_kernel<false, 3>
        <<<(M / BM) * (D / BN), 256, 0, stream>>>(e_b, We_b, nullptr, he_b, heT_b, M, D, D, 1.0f);

    // hn[N,D] = x @ Wn^T   -> bf16
    gemm_wmma_kernel<false, 1>
        <<<(N / BM) * (D / BN), 256, 0, stream>>>(x_b, Wn_b, nullptr, hn_b, nullptr, N, D, D, 1.0f);

    // S[N,M] = hn @ he^T   -> fp32 directly into d_out's B region
    gemm_wmma_kernel<false, 0>
        <<<(N / BM) * (M / BN), 256, 0, stream>>>(hn_b, he_b, Bmat, nullptr, nullptr, N, M, D, 1.0f);

    // masked softmax in place, scale = 1/sqrt(1024) = 1/32
    softmax_mask_kernel<<<N, 256, 0, stream>>>(Bmat, H, 0.03125f);

    // out[N,D] = elu( B[N,M] @ heT[D,M]^T )  (A fp32, converted during staging)
    gemm_wmma_kernel<true, 2>
        <<<(N / BM) * (D / BN), 256, 0, stream>>>(Bmat, heT_b, outp, nullptr, nullptr, N, D, M, 1.0f);
}
